// EntityRelationHead_84902913507933
// MI455X (gfx1250) — matrix operand order repaired and checked by reading, not compiled, not run on hardware
//
#include <hip/hip_runtime.h>

#define Hdim 768
#define Ldim 512
#define Bdim 2
#define Rdim 32
#define Edim 9

typedef __attribute__((ext_vector_type(16))) _Float16 v16h;
typedef __attribute__((ext_vector_type(8)))  _Float16 v8h;
typedef __attribute__((ext_vector_type(8)))  float    v8f;

__device__ __forceinline__ v16h combine8(v8h lo, v8h hi) {
  return __builtin_shufflevector(lo, hi, 0, 1, 2, 3, 4, 5, 6, 7,
                                 8, 9, 10, 11, 12, 13, 14, 15);
}

// A-operand (16x32 f16, ISA 7.12.2) from a row-major f16 row pointer:
// lane half h: elements [0..15] = K 16h..16h+15 (contiguous).
__device__ __forceinline__ v16h load_A_row(const _Float16* row, int half) {
  const v8h lo = *(const v8h*)(row + 16 * half);
  const v8h hi = *(const v8h*)(row + 16 * half + 8);
  return combine8(lo, hi);
}

// ---------------------------------------------------------------------------
// Kernel 1: hp = relu(h @ proj_W^T + proj_b) -> f16 workspace. 64 M-tiles,
// 8 waves each owning 6 N-tiles of 16 (48*16 = 768 cols).
// ---------------------------------------------------------------------------
__global__ __launch_bounds__(256) void proj_kernel(
    const float* __restrict__ h, const float* __restrict__ W,
    const float* __restrict__ bias, _Float16* __restrict__ hp) {
  __shared__ _Float16 As[16 * 32];
  const int mt   = blockIdx.x;
  const int tid  = threadIdx.x;
  const int lane = tid & 31;
  const int w    = tid >> 5;
  const int m16  = lane & 15;
  const int half = (lane >> 4) & 1;

  v8f acc[6];
#pragma unroll
  for (int q = 0; q < 6; ++q) acc[q] = (v8f)0.f;

  for (int k0 = 0; k0 < Hdim; k0 += 32) {
    {  // stage A tile (16x32 f32 -> f16), 2 elems/thread
      const int idx = tid * 2;
      const int m = idx >> 5, k = idx & 31;
      const float2 hv =
          *(const float2*)(h + (size_t)(mt * 16 + m) * Hdim + k0 + k);
      As[idx]     = (_Float16)hv.x;
      As[idx + 1] = (_Float16)hv.y;
    }
    __syncthreads();
    const v16h a = load_A_row(As + m16 * 32, half);
#pragma unroll
    for (int q = 0; q < 6; ++q) {
      // B[k][n] = proj_W[n][k]; contiguous K 16h..16h+15 in row n
      const int col = (w * 6 + q) * 16 + m16;
      const float* wr = W + (size_t)col * Hdim + k0 + 16 * half;
      v16h bm;
#pragma unroll
      for (int p = 0; p < 8; ++p) {
        const float2 wv = *(const float2*)(wr + 2 * p);
        bm[2 * p]     = (_Float16)wv.x;
        bm[2 * p + 1] = (_Float16)wv.y;
      }
      acc[q] = __builtin_amdgcn_wmma_f32_16x16x32_f16(
          false, a, false, bm, (short)0, acc[q], false, false);
    }
    __syncthreads();
  }

#pragma unroll
  for (int q = 0; q < 6; ++q) {
    const int col = (w * 6 + q) * 16 + m16;
    const float bv = bias[col];
#pragma unroll
    for (int v = 0; v < 8; ++v) {
      const int row = mt * 16 + v + 8 * half;
      float val = acc[q][v] + bv;
      val = val > 0.f ? val : 0.f;
      hp[(size_t)row * Hdim + col] = (_Float16)val;
    }
  }
}

// ---------------------------------------------------------------------------
// Kernel 2: ent_logits = hp @ ent_W^T + ent_b (N=9, negligible).
// ---------------------------------------------------------------------------
__global__ __launch_bounds__(256) void ent_kernel(
    const _Float16* __restrict__ hp, const float* __restrict__ entW,
    const float* __restrict__ entb, float* __restrict__ out) {
  const int t = blockIdx.x * blockDim.x + threadIdx.x;
  if (t >= Bdim * Ldim * Edim) return;
  const int row = t / Edim, e = t - row * Edim;
  const _Float16* hrow = hp + (size_t)row * Hdim;
  const float* wrow = entW + (size_t)e * Hdim;
  float acc = 0.f;
  for (int k = 0; k < Hdim; ++k) acc += (float)hrow[k] * wrow[k];
  out[t] = acc + entb[e];
}

// ---------------------------------------------------------------------------
// Kernel 3: rel[b,i,j,r] = (hp[b,i] @ bil_W[r]) . hp[b,j] + bil_b[r]
// One WG per (r, 32-row i-tile, b): 1024 WGs. Phase A builds a 32x768 f16
// t-tile in LDS (each bil_W element feeds 2 WMMAs); phase B contracts it
// against hp[b,j,:]^T with single-v16h B-operand loads.
// ---------------------------------------------------------------------------
__global__ __launch_bounds__(256) void rel_kernel(
    const _Float16* __restrict__ hp, const float* __restrict__ bilW,
    const float* __restrict__ bilb, float* __restrict__ rel) {
  __shared__ _Float16 Ts[32 * Hdim];  // 48 KB t-tile (rows = 32 i's)
  __shared__ _Float16 As[32 * 32];    // 2 KB staged hp slab
  const int r    = blockIdx.x;
  const int it   = blockIdx.y;  // 0..15, 32 rows each
  const int bt   = blockIdx.z;
  const int tid  = threadIdx.x;
  const int lane = tid & 31;
  const int w    = tid >> 5;
  const int m16  = lane & 15;
  const int half = (lane >> 4) & 1;

  // ---- Phase A: Ts = hp_tile(32x768) @ bil_W[r](768x768) ----
  v8f acc[2][6];
#pragma unroll
  for (int ms = 0; ms < 2; ++ms)
#pragma unroll
    for (int q = 0; q < 6; ++q) acc[ms][q] = (v8f)0.f;
  const float* Wr = bilW + (size_t)r * Hdim * Hdim;

  for (int k0 = 0; k0 < Hdim; k0 += 32) {
    {  // stage 32x32 hp slab (f16), 4 elems (8B) per thread
      const int idx = tid * 4;
      const int m = idx >> 5, k = idx & 31;
      const int row = bt * Ldim + it * 32 + m;
      *(uint64_t*)&As[idx] =
          *(const uint64_t*)(hp + (size_t)row * Hdim + k0 + k);
    }
    __syncthreads();
    const v16h a0 = load_A_row(As + m16 * 32, half);
    const v16h a1 = load_A_row(As + (16 + m16) * 32, half);
#pragma unroll
    for (int q = 0; q < 6; ++q) {
      // B[k][n] = bil_W[r][k][n], K-major rows 768 floats apart
      const int col = (w * 6 + q) * 16 + m16;
      const float* wp = Wr + (size_t)(k0 + 16 * half) * Hdim + col;
      v16h bm;
#pragma unroll
      for (int p = 0; p < 8; ++p) {
        bm[2 * p]     = (_Float16)wp[(size_t)(2 * p) * Hdim];
        bm[2 * p + 1] = (_Float16)wp[(size_t)(2 * p + 1) * Hdim];
      }
      acc[0][q] = __builtin_amdgcn_wmma_f32_16x16x32_f16(
          false, a0, false, bm, (short)0, acc[0][q], false, false);
      acc[1][q] = __builtin_amdgcn_wmma_f32_16x16x32_f16(
          false, a1, false, bm, (short)0, acc[1][q], false, false);
    }
    __syncthreads();
  }
#pragma unroll
  for (int ms = 0; ms < 2; ++ms)
#pragma unroll
    for (int q = 0; q < 6; ++q) {
      const int col = (w * 6 + q) * 16 + m16;
#pragma unroll
      for (int v = 0; v < 8; ++v)
        Ts[(ms * 16 + v + 8 * half) * Hdim + col] = (_Float16)acc[ms][q][v];
    }
  __syncthreads();

  // ---- Phase B: rel tile (32 x 512) for this (b, it, r) ----
  const float bv = bilb[r];
#pragma unroll
  for (int jj = 0; jj < 4; ++jj) {
    const int jt = w * 4 + jj;
    const int jrow = bt * Ldim + jt * 16 + m16;  // lane%16 = N column = j
    const _Float16* hj = hp + (size_t)jrow * Hdim + 16 * half;
    v8f c0 = (v8f)0.f, c1 = (v8f)0.f;
    for (int k0 = 0; k0 < Hdim; k0 += 32) {
      const v16h bm = *(const v16h*)(hj + k0);  // one 32B contiguous load
      const v16h a0 = load_A_row(Ts + m16 * Hdim + k0, half);
      const v16h a1 = load_A_row(Ts + (16 + m16) * Hdim + k0, half);
      c0 = __builtin_amdgcn_wmma_f32_16x16x32_f16(
          false, a0, false, bm, (short)0, c0, false, false);
      c1 = __builtin_amdgcn_wmma_f32_16x16x32_f16(
          false, a1, false, bm, (short)0, c1, false, false);
    }
#pragma unroll
    for (int v = 0; v < 8; ++v) {
      const int j = jt * 16 + m16;
      const int i0 = it * 32 + v + 8 * half;
      const int i1 = i0 + 16;
      rel[(((size_t)bt * Ldim + i0) * Ldim + j) * Rdim + r] = c0[v] + bv;
      rel[(((size_t)bt * Ldim + i1) * Ldim + j) * Rdim + r] = c1[v] + bv;
    }
  }
}

// ---------------------------------------------------------------------------
extern "C" void kernel_launch(void* const* d_in, const int* in_sizes, int n_in,
                              void* d_out, int out_size, void* d_ws,
                              size_t ws_size, hipStream_t stream) {
  (void)in_sizes; (void)n_in; (void)out_size; (void)ws_size;
  const float* h     = (const float*)d_in[0];
  const float* projW = (const float*)d_in[1];
  const float* projb = (const float*)d_in[2];
  const float* entW  = (const float*)d_in[3];
  const float* entb  = (const float*)d_in[4];
  const float* bilW  = (const float*)d_in[5];
  const float* bilb  = (const float*)d_in[6];

  float* out_ent = (float*)d_out;
  float* out_rel = out_ent + (size_t)Bdim * Ldim * Edim;
  _Float16* hp = (_Float16*)d_ws;  // 1024*768 f16 = 1.5 MB scratch

  proj_kernel<<<dim3(Bdim * Ldim / 16), 256, 0, stream>>>(h, projW, projb, hp);
  ent_kernel<<<dim3((Bdim * Ldim * Edim + 255) / 256), 256, 0, stream>>>(
      hp, entW, entb, out_ent);
  rel_kernel<<<dim3(Rdim, Ldim / 32, Bdim), 256, 0, stream>>>(
      hp, bilW, bilb, out_rel);
}
